// Linear_33088428048386
// MI455X (gfx1250) — matrix operand order repaired and checked by reading, not compiled, not run on hardware
//
#include <hip/hip_runtime.h>
#include <hip/hip_bf16.h>
#include <stdint.h>

// ---- Problem constants (must match reference) ----
#define NXD 256
#define NUD 128
#define NYD 128
#define NBD 32
#define NTD 4096

typedef __attribute__((ext_vector_type(16))) _Float16 v16h;
typedef __attribute__((ext_vector_type(8)))  float    v8f;

// Vector-of-4-int type matching the async builtin's parameter type
// ('__attribute__((__vector_size__(4 * sizeof(int)))) int').
typedef int vi4 __attribute__((vector_size(4 * sizeof(int))));
typedef __attribute__((address_space(1))) vi4 gvi4;   // global (device) b128
typedef __attribute__((address_space(3))) vi4 lvi4;   // LDS b128

struct __align__(16) H8 { _Float16 h[8]; };

__device__ __forceinline__ v8f wmma_f16(v16h a, v16h b, v8f c) {
    // D = A(16x32) * B(32x16) + C, f32 accumulate
    return __builtin_amdgcn_wmma_f32_16x16x32_f16(
        /*neg_a=*/false, a, /*neg_b=*/false, b,
        /*c_mod=*/(short)0, c, /*reuse_a=*/false, /*reuse_b=*/false);
}

// ---- CDNA5 async LDS->global store (ASYNCcnt-tracked DMA) ----------------
__device__ __forceinline__ void async_store_b128(void* gptr, void* lptr) {
#if __has_builtin(__builtin_amdgcn_global_store_async_from_lds_b128)
    __builtin_amdgcn_global_store_async_from_lds_b128(
        (gvi4*)gptr, (lvi4*)lptr, 0, 0);
#else
    unsigned long long ga = (unsigned long long)(uintptr_t)gptr;
    unsigned la = (unsigned)(uintptr_t)(__attribute__((address_space(3))) void*)lptr;
    asm volatile("global_store_async_from_lds_b128 %0, %1, off"
                 :: "v"(ga), "v"(la) : "memory");
#endif
}

__device__ __forceinline__ void wait_asynccnt_le2() {
#if __has_builtin(__builtin_amdgcn_s_wait_asynccnt)
    __builtin_amdgcn_s_wait_asynccnt(2);
#else
    asm volatile("s_wait_asynccnt 0x2" ::: "memory");
#endif
}

// ---------------------------------------------------------------------------
// Prep: pack a row-major f32 matrix [mtiles*16, ktiles*32] into f16 WMMA
// A-operand tile layout: tile(mt,kc) = 32 lanes x 16 halves (32B/lane).
//   lane l (<16): M = mt*16+l,    K = kc*32 + i
//   lane l (>=16):M = mt*16+l-16, K = kc*32 + 16 + i
// ---------------------------------------------------------------------------
__global__ void pack_a_operand(const float* __restrict__ src,
                               _Float16* __restrict__ dst,
                               int ktiles, int total) {
    int id = blockIdx.x * blockDim.x + threadIdx.x;
    if (id >= total) return;
    int i    = id & 15;
    int l    = (id >> 4) & 31;
    int tile = id >> 9;
    int kc   = tile % ktiles;
    int mt   = tile / ktiles;
    int M    = mt * 16 + (l & 15);
    int Koff = ((l >> 4) << 4) + i;
    int K    = kc * 32 + Koff;
    int cols = ktiles * 32;
    dst[id] = (_Float16)src[(size_t)M * cols + K];
}

// ---------------------------------------------------------------------------
// Prep: pack us[NB][NT][NU] f32 into f16 B-operand tiles:
//   UT[t][ntile][kc(4)][lane(32)][16]  (one 32B chunk per thread)
//   lane l: N = ntile*16 + (l&15), K = kc*32 + (l>=16)*16 + i
// ---------------------------------------------------------------------------
__global__ void pack_us(const float* __restrict__ us, _Float16* __restrict__ dst) {
    int id = blockIdx.x * blockDim.x + threadIdx.x;   // 1,048,576 chunks
    int lane  = id & 31;
    int kc    = (id >> 5) & 3;
    int ntile = (id >> 7) & 1;
    int t     = id >> 8;
    int n     = ntile * 16 + (lane & 15);
    int Kb    = kc * 32 + ((lane >> 4) << 4);
    const float* s = us + ((size_t)n * NTD + t) * NUD + Kb;
    _Float16*    d = dst + (size_t)id * 16;
#pragma unroll
    for (int i = 0; i < 16; ++i) d[i] = (_Float16)s[i];
}

// ---------------------------------------------------------------------------
// Phase 1: sequential recurrence. 2 blocks (one per 16-column batch tile),
// 256 threads = 8 wave32, 2 M-tiles per wave. A,B register-resident;
// X rotates through 3 LDS buffers in B-operand layout; pre-update state is
// archived to Xpk with async LDS->global stores (ASYNCcnt).
// ---------------------------------------------------------------------------
__global__ __launch_bounds__(256, 1)
void ssm_recurrence(const float* __restrict__ x0,
                    const _Float16* __restrict__ Apk,
                    const _Float16* __restrict__ Bpk,
                    const _Float16* __restrict__ UT,
                    _Float16* __restrict__ Xpk) {
    extern __shared__ _Float16 smem[];      // 3 * 4096 halves (3 x 8KB)
    _Float16* xtA = smem;
    _Float16* xtB = smem + 4096;
    _Float16* xtC = smem + 8192;

    const int tid   = threadIdx.x;
    const int lane  = tid & 31;
    const int wave  = tid >> 5;             // 0..7
    const int ntile = blockIdx.x;           // 0..1

    // Register-resident A (16x256) and B (16x128) row blocks for our 2 M-tiles
    v16h Areg[2][8];
    v16h Breg[2][4];
#pragma unroll
    for (int m2 = 0; m2 < 2; ++m2) {
        const int mt = wave * 2 + m2;
#pragma unroll
        for (int kc = 0; kc < 8; ++kc)
            Areg[m2][kc] = *(const v16h*)(Apk + (size_t)(mt * 8 + kc) * 512 + lane * 16);
#pragma unroll
        for (int kc = 0; kc < 4; ++kc)
            Breg[m2][kc] = *(const v16h*)(Bpk + (size_t)(mt * 4 + kc) * 512 + lane * 16);
    }

    // Initialize X^T LDS (B-operand layout) from x0 (f32, [NB][NX])
    {
        const int n  = tid >> 4;            // 0..15 batch col within tile
        const int kb = (tid & 15) << 4;     // K base, multiple of 16
        const float* s = x0 + (size_t)(ntile * 16 + n) * NXD + kb;
        const int kc   = kb >> 5;
        const int slot = ((kb & 31) < 16) ? n : (16 + n);
        _Float16* d = xtA + kc * 512 + slot * 16;
#pragma unroll
        for (int i = 0; i < 16; ++i) d[i] = (_Float16)s[i];
    }
    __syncthreads();

    _Float16* cur = xtA;
    _Float16* nxt = xtB;
    _Float16* thr = xtC;
    for (int t = 0; t < NTD; ++t) {
        // Archive pre-update state (already in B-operand layout) for phase 2:
        // async DMA straight out of LDS, no VGPR round-trip.
        {
            char* g = (char*)Xpk + (size_t)(t * 2 + ntile) * 8192 + (size_t)tid * 32;
            char* l = (char*)cur + tid * 32;
            async_store_b128(g,      l);
            async_store_b128(g + 16, l + 16);
        }

        // B-operand X tiles (shared across both M-tiles of this wave)
        v16h xreg[8];
#pragma unroll
        for (int kc = 0; kc < 8; ++kc)
            xreg[kc] = *(const v16h*)(cur + kc * 512 + lane * 16);

        // x_next = A @ x
        v8f acc0 = {}, acc1 = {};
#pragma unroll
        for (int kc = 0; kc < 8; ++kc) {
            acc0 = wmma_f16(Areg[0][kc], xreg[kc], acc0);
            acc1 = wmma_f16(Areg[1][kc], xreg[kc], acc1);
        }
        //        + B @ u   (u tiles streamed from pre-packed global)
        const _Float16* ut = UT + (size_t)(t * 2 + ntile) * 2048;
#pragma unroll
        for (int kc = 0; kc < 4; ++kc) {
            v16h u = *(const v16h*)(ut + kc * 512 + lane * 16);
            acc0 = wmma_f16(Breg[0][kc], u, acc0);
            acc1 = wmma_f16(Breg[1][kc], u, acc1);
        }

        // Scatter accumulators into next X^T (B-operand layout): one aligned
        // 16B LDS store per tile per lane.
        const int n  = lane & 15;
        const int hi = lane >> 4;
#pragma unroll
        for (int m2 = 0; m2 < 2; ++m2) {
            const int mt = wave * 2 + m2;
            const v8f a  = m2 ? acc1 : acc0;
            H8 p;
#pragma unroll
            for (int v = 0; v < 8; ++v) p.h[v] = (_Float16)a[v];
            _Float16* d = nxt + (mt >> 1) * 512 + ((mt & 1) ? (16 + n) : n) * 16 + hi * 8;
            *(H8*)d = p;
        }

        // Async stores issued at step t-1 read buffer `thr` (next write target
        // after rotation). Triple buffering + this wait ensures they finished
        // before anyone rewrites it (stores complete in order).
        wait_asynccnt_le2();
        __syncthreads();
        _Float16* tmp = cur; cur = nxt; nxt = thr; thr = tmp;
    }
}

// ---------------------------------------------------------------------------
// Phase 2: y_t = C x_t + D u_t, fully parallel over (t, batch-tile).
// 512 blocks x 256 threads; each wave owns one NY M-tile, 16 timesteps/block.
// ---------------------------------------------------------------------------
__global__ __launch_bounds__(256, 1)
void ssm_output(const _Float16* __restrict__ Cpk,
                const _Float16* __restrict__ Dpk,
                const _Float16* __restrict__ Xpk,
                const _Float16* __restrict__ UT,
                float* __restrict__ out) {
    const int tid   = threadIdx.x;
    const int lane  = tid & 31;
    const int mt    = tid >> 5;                 // 0..7 (NY tiles)
    const int ntile = blockIdx.x & 1;
    const int tbase = (blockIdx.x >> 1) << 4;   // 16 timesteps per block

    v16h Creg[8];
    v16h Dreg[4];
#pragma unroll
    for (int kc = 0; kc < 8; ++kc)
        Creg[kc] = *(const v16h*)(Cpk + (size_t)(mt * 8 + kc) * 512 + lane * 16);
#pragma unroll
    for (int kc = 0; kc < 4; ++kc)
        Dreg[kc] = *(const v16h*)(Dpk + (size_t)(mt * 4 + kc) * 512 + lane * 16);

    const int n  = ntile * 16 + (lane & 15);
    const int hi = lane >> 4;

    for (int tt = 0; tt < 16; ++tt) {
        const int t = tbase + tt;
        const _Float16* xp = Xpk + (size_t)(t * 2 + ntile) * 4096;
        const _Float16* up = UT  + (size_t)(t * 2 + ntile) * 2048;
        v8f acc = {};
#pragma unroll
        for (int kc = 0; kc < 8; ++kc)
            acc = wmma_f16(Creg[kc], *(const v16h*)(xp + kc * 512 + lane * 16), acc);
#pragma unroll
        for (int kc = 0; kc < 4; ++kc)
            acc = wmma_f16(Dreg[kc], *(const v16h*)(up + kc * 512 + lane * 16), acc);

        // out[NB][NT][NY]: 8 consecutive M per lane -> two float4 stores
        float* o = out + ((size_t)n * NTD + t) * NYD + mt * 16 + hi * 8;
        float4 lo4 = make_float4(acc[0], acc[1], acc[2], acc[3]);
        float4 hi4 = make_float4(acc[4], acc[5], acc[6], acc[7]);
        *(float4*)(o)     = lo4;
        *(float4*)(o + 4) = hi4;
    }
}

// ---------------------------------------------------------------------------
// Workspace layout (bytes):
//   Apk 128KB | Bpk 64KB | Cpk 64KB | Dpk 32KB | UT 32MB | Xpk 64MB
// ---------------------------------------------------------------------------
#define WS_APK 0
#define WS_BPK 131072
#define WS_CPK 196608
#define WS_DPK 262144
#define WS_UT  294912
#define WS_XPK (294912 + 33554432)

extern "C" void kernel_launch(void* const* d_in, const int* in_sizes, int n_in,
                              void* d_out, int out_size, void* d_ws, size_t ws_size,
                              hipStream_t stream) {
    const float* x0 = (const float*)d_in[0];
    const float* us = (const float*)d_in[1];
    const float* A  = (const float*)d_in[2];
    const float* B  = (const float*)d_in[3];
    const float* C  = (const float*)d_in[4];
    const float* D  = (const float*)d_in[5];
    float* out = (float*)d_out;

    char* ws = (char*)d_ws;
    _Float16* Apk = (_Float16*)(ws + WS_APK);
    _Float16* Bpk = (_Float16*)(ws + WS_BPK);
    _Float16* Cpk = (_Float16*)(ws + WS_CPK);
    _Float16* Dpk = (_Float16*)(ws + WS_DPK);
    _Float16* UT  = (_Float16*)(ws + WS_UT);
    _Float16* Xpk = (_Float16*)(ws + WS_XPK);

    // Pack weights into WMMA A-operand tile layouts (f16)
    pack_a_operand<<<256, 256, 0, stream>>>(A, Apk, 8, 65536);  // 16x8 tiles
    pack_a_operand<<<128, 256, 0, stream>>>(B, Bpk, 4, 32768);  // 16x4 tiles
    pack_a_operand<<<128, 256, 0, stream>>>(C, Cpk, 8, 32768);  //  8x8 tiles
    pack_a_operand<<< 64, 256, 0, stream>>>(D, Dpk, 4, 16384);  //  8x4 tiles

    // Pack inputs into B-operand tiles (f16)
    pack_us<<<4096, 256, 0, stream>>>(us, UT);

    // Phase 1: sequential recurrence (2 batch tiles in parallel)
    ssm_recurrence<<<2, 256, 3 * 4096 * sizeof(_Float16), stream>>>(x0, Apk, Bpk, UT, Xpk);

    // Phase 2: outputs, fully parallel
    ssm_output<<<(NTD / 16) * 2, 256, 0, stream>>>(Cpk, Dpk, Xpk, UT, out);
}
